// CausalSelfAttention_36455682409076
// MI455X (gfx1250) — matrix operand order repaired and checked by reading, not compiled, not run on hardware
//
#include <hip/hip_runtime.h>
#include <hip/hip_bf16.h>

// ---------------------------------------------------------------------------
// MLA causal attention pipeline for gfx1250 (MI455X).
// bf16 WMMA everywhere; GEMMs double-buffer via ASYNC global->LDS loads;
// attention K-tiles stream via the Tensor Data Mover (TENSORcnt).
// ---------------------------------------------------------------------------

typedef __bf16 bf16;
typedef __bf16 v16bf __attribute__((ext_vector_type(16)));
typedef float  v8f   __attribute__((ext_vector_type(8)));
typedef unsigned int u32x4 __attribute__((ext_vector_type(4)));
typedef int          i32x4 __attribute__((ext_vector_type(4)));
typedef int          i32x8 __attribute__((ext_vector_type(8)));

union Frag { u32x4 u[2]; v16bf v; };   // 32 bytes = one 16-bit WMMA operand/lane

__device__ __forceinline__ v8f v8f_zero() {
    v8f z;
#pragma unroll
    for (int r = 0; r < 8; ++r) z[r] = 0.0f;
    return z;
}

// ---- ASYNC global->LDS 16B copy (per-lane addresses, ASYNCcnt) ----
__device__ __forceinline__ void async_copy_b128(void* lds_dst, const void* gsrc) {
    unsigned lo = (unsigned)(unsigned long long)lds_dst;  // low 32b = LDS offset
    asm volatile("global_load_async_to_lds_b128 %0, %1, off"
                 :: "v"(lo), "v"(gsrc) : "memory");
}
__device__ __forceinline__ void wait_async_le4() {
    asm volatile("s_wait_asynccnt 0x4" ::: "memory");
}
__device__ __forceinline__ void wait_async_0() {
    asm volatile("s_wait_asynccnt 0x0" ::: "memory");
}

// ---- Tensor Data Mover: 2D bf16 tile (rows x 576) from Kg into LDS ----
// D# per CDNA5 ISA ch.8: group0 = {flags, lds_addr, global_addr, type=2},
// group1 = dims/strides (data_size units), groups 2/3 zero (2-D tensor).
__device__ __forceinline__ void tdm_load_k_tile(const bf16* gsrc, void* lds_dst) {
    unsigned long long ga = (unsigned long long)gsrc;
    u32x4 g0;
    g0[0] = 1u;                                             // count=1 (valid user D#)
    g0[1] = (unsigned)(unsigned long long)lds_dst;          // lds_addr (bytes)
    g0[2] = (unsigned)(ga & 0xffffffffu);                   // global_addr[31:0]
    g0[3] = (unsigned)((ga >> 32) & 0x01ffffffu) | (2u << 30);  // addr[56:32] | type=2
    i32x8 g1;
    g1[0] = 0x00010000;                 // workgroup_mask=0, data_size=1 (2 bytes)
    g1[1] = (int)(576u << 16);          // tensor_dim0[15:0]=576 (<<16); barrier addr=0
    g1[2] = (int)(2048u << 16);         // tensor_dim0 hi=0 | tensor_dim1[15:0]=2048
    g1[3] = (int)(576u << 16);          // tensor_dim1 hi=0 | tile_dim0=576
    g1[4] = 64;                         // tile_dim1=64 | tile_dim2=0
    g1[5] = 576;                        // tensor_dim0_stride[31:0] (elements)
    g1[6] = 0;                          // stride0 hi | tensor_dim1_stride lo
    g1[7] = 0;
    i32x4 z4 = {0, 0, 0, 0};
#if __clang_major__ >= 23
    i32x8 z8 = {0, 0, 0, 0, 0, 0, 0, 0};
    __builtin_amdgcn_tensor_load_to_lds(g0, g1, z4, z4, z8, 0);
#else
    __builtin_amdgcn_tensor_load_to_lds(g0, g1, z4, z4, 0);
#endif
}

// Model dimensions
#define TT     2048
#define DEMB   2048
#define NHEAD  16
#define KVLOW  512
#define ROPE   64
#define QKHS   576      // KVLOW + ROPE
#define QLOW   1024
#define NQKV   1600     // KVLOW + ROPE + QLOW
#define QDIM   9216     // NHEAD * QKHS
#define YDIM   8192     // NHEAD * KVLOW

// ---------------------------------------------------------------------------
// f32 -> bf16 (row-major, grid-stride)
// ---------------------------------------------------------------------------
__global__ void cvt_bf16_kernel(const float* __restrict__ in, bf16* __restrict__ out, int n) {
    for (int i = blockIdx.x * blockDim.x + threadIdx.x; i < n; i += gridDim.x * blockDim.x)
        out[i] = (bf16)in[i];
}

// ---------------------------------------------------------------------------
// f32[R][C] -> bf16[C][R] tiled transpose-convert (R,C multiples of 32)
// ---------------------------------------------------------------------------
__global__ __launch_bounds__(256) void cvt_bf16_t_kernel(
    const float* __restrict__ in, bf16* __restrict__ out, int R, int C)
{
    __shared__ bf16 tile[32][33];
    const int c0 = blockIdx.x * 32;
    const int r0 = blockIdx.y * 32;
    const int tx = threadIdx.x & 31, ty = threadIdx.x >> 5;  // 32x8
#pragma unroll
    for (int i = 0; i < 32; i += 8)
        tile[ty + i][tx] = (bf16)in[(size_t)(r0 + ty + i) * C + c0 + tx];
    __syncthreads();
#pragma unroll
    for (int i = 0; i < 32; i += 8)
        out[(size_t)(c0 + ty + i) * R + r0 + tx] = tile[tx][ty + i];
}

// ---------------------------------------------------------------------------
// bf16[R][Cstride] (first blocks of cols) -> bf16[C][R] transpose
// ---------------------------------------------------------------------------
__global__ __launch_bounds__(256) void transpose_bf16_kernel(
    const bf16* __restrict__ in, bf16* __restrict__ out, int R, int Cstride)
{
    __shared__ bf16 tile[32][33];
    const int c0 = blockIdx.x * 32;
    const int r0 = blockIdx.y * 32;
    const int tx = threadIdx.x & 31, ty = threadIdx.x >> 5;
#pragma unroll
    for (int i = 0; i < 32; i += 8)
        tile[ty + i][tx] = in[(size_t)(r0 + ty + i) * Cstride + c0 + tx];
    __syncthreads();
#pragma unroll
    for (int i = 0; i < 32; i += 8)
        out[(size_t)(c0 + ty + i) * R + r0 + tx] = tile[tx][ty + i];
}

// ---------------------------------------------------------------------------
// GEMM (TN): C[M,N] = A[M,K] * Bt[N,K]^T, bf16 in, f32 accumulate.
// 256 threads = 8 waves, tile 128x128x32, double-buffered ASYNC LDS staging.
// Steady-state k-loop is branch-free (last iteration peeled).
// ---------------------------------------------------------------------------
#define BM 128
#define BN 128
#define BK 32

__global__ __launch_bounds__(256) void gemm_bf16_tn_kernel(
    const bf16* __restrict__ A, const bf16* __restrict__ Bt, float* __restrict__ C,
    int M, int N, int K)
{
    __shared__ __align__(16) bf16 As[2][BM * BK];   // [m][k]
    __shared__ __align__(16) bf16 Bs[2][BN * BK];   // [n][k]

    const int tid  = threadIdx.x;
    const int wave = tid >> 5;
    const int lane = tid & 31;
    const int hf   = lane >> 4;
    const int l16  = lane & 15;
    const int wm   = wave >> 1;     // 0..3 : 32-row slab
    const int wn   = wave & 1;      // 0..1 : 64-col slab

    const int block_m = blockIdx.y * BM;
    const int block_n = blockIdx.x * BN;

    // this thread's two 16B staging chunks (same pattern for A and B)
    const int c0row = tid >> 2,           c0cg = (tid & 3) * 8;
    const int c1row = (tid + 256) >> 2,   c1cg = ((tid + 256) & 3) * 8;
    const int bn0 = (block_n + c0row < N) ? block_n + c0row : N - 1;
    const int bn1 = (block_n + c1row < N) ? block_n + c1row : N - 1;

    auto stage = [&](int bufi, int k0) {
        async_copy_b128(&As[bufi][c0row * BK + c0cg],
                        A + (size_t)(block_m + c0row) * K + k0 + c0cg);
        async_copy_b128(&As[bufi][c1row * BK + c1cg],
                        A + (size_t)(block_m + c1row) * K + k0 + c1cg);
        async_copy_b128(&Bs[bufi][c0row * BK + c0cg],
                        Bt + (size_t)bn0 * K + k0 + c0cg);
        async_copy_b128(&Bs[bufi][c1row * BK + c1cg],
                        Bt + (size_t)bn1 * K + k0 + c1cg);
    };

    v8f acc[2][4];
#pragma unroll
    for (int i = 0; i < 2; ++i)
#pragma unroll
        for (int j = 0; j < 4; ++j) acc[i][j] = v8f_zero();

    auto compute = [&](int buf) {
        Frag af[2], bfg[4];
#pragma unroll
        for (int mt = 0; mt < 2; ++mt) {
            const bf16* p = &As[buf][(wm * 32 + mt * 16 + l16) * BK + hf * 8];
            af[mt].u[0] = *(const u32x4*)p;
            af[mt].u[1] = *(const u32x4*)(p + 16);
        }
#pragma unroll
        for (int nt = 0; nt < 4; ++nt) {
            const bf16* p = &Bs[buf][(wn * 64 + nt * 16 + l16) * BK + hf * 8];
            bfg[nt].u[0] = *(const u32x4*)p;
            bfg[nt].u[1] = *(const u32x4*)(p + 16);
        }
#pragma unroll
        for (int mt = 0; mt < 2; ++mt)
#pragma unroll
            for (int nt = 0; nt < 4; ++nt)
                acc[mt][nt] = __builtin_amdgcn_wmma_f32_16x16x32_bf16(
                    false, af[mt].v, false, bfg[nt].v, (short)0, acc[mt][nt], false, false);
    };

    const int nk = K / BK;
    stage(0, 0);
    for (int s = 0; s < nk - 1; ++s) {          // branch-free steady state
        stage((s & 1) ^ 1, (s + 1) * BK);
        wait_async_le4();
        __syncthreads();
        compute(s & 1);
        __syncthreads();
    }
    wait_async_0();
    __syncthreads();
    compute((nk - 1) & 1);

#pragma unroll
    for (int mt = 0; mt < 2; ++mt) {
        int rbase = block_m + wm * 32 + mt * 16 + hf * 8;
#pragma unroll
        for (int nt = 0; nt < 4; ++nt) {
            int col = block_n + wn * 64 + nt * 16 + l16;
            if (col < N) {
#pragma unroll
                for (int r = 0; r < 8; ++r)
                    C[(size_t)(rbase + r) * N + col] = acc[mt][nt][r];
            }
        }
    }
}

// ---------------------------------------------------------------------------
// Split latents -> K (RoPE on last 64 dims) bf16 and c_q bf16
// ---------------------------------------------------------------------------
__global__ void split_rope_kernel(const float* __restrict__ lat,
                                  const float* __restrict__ cosT,
                                  const float* __restrict__ sinT,
                                  bf16* __restrict__ Kg, bf16* __restrict__ Cq)
{
    const int t = blockIdx.x;
    const float* row = lat + (size_t)t * NQKV;
    for (int i = threadIdx.x; i < NQKV; i += blockDim.x) {
        if (i < KVLOW) {
            Kg[t * QKHS + i] = (bf16)row[i];
        } else if (i < QKHS) {
            int j = i - KVLOW;
            float x   = row[i];
            float rot = (j < 32) ? -row[KVLOW + j + 32] : row[KVLOW + j - 32];
            Kg[t * QKHS + i] = (bf16)(x * cosT[t * ROPE + j] + rot * sinT[t * ROPE + j]);
        } else {
            Cq[t * QLOW + (i - QKHS)] = (bf16)row[i];
        }
    }
}

// ---------------------------------------------------------------------------
// q [T, NH*576] f32 -> Qh [NH, T, 576] bf16 with RoPE on last 64 of each head
// ---------------------------------------------------------------------------
__global__ void q_rope_kernel(const float* __restrict__ qf,
                              const float* __restrict__ cosT,
                              const float* __restrict__ sinT,
                              bf16* __restrict__ Qh)
{
    const int t = blockIdx.x;
    const float* row = qf + (size_t)t * QDIM;
    for (int i = threadIdx.x; i < QDIM; i += blockDim.x) {
        int hd = i / QKHS, d = i % QKHS;
        float v;
        if (d < KVLOW) {
            v = row[i];
        } else {
            int j = d - KVLOW;
            const float* qr = row + hd * QKHS + KVLOW;
            float x   = qr[j];
            float rot = (j < 32) ? -qr[j + 32] : qr[j - 32];
            v = x * cosT[t * ROPE + j] + rot * sinT[t * ROPE + j];
        }
        Qh[((size_t)hd * TT + t) * QKHS + d] = (bf16)v;
    }
}

// ---------------------------------------------------------------------------
// Flash causal attention: block = (head, 32 q rows); key tiles of 64.
// K tiles (64x576 bf16, 72KB) stream via TDM tensor_load_to_lds, double
// buffered on TENSORcnt; V^T fragments read from pre-transposed Kt (L2).
// ---------------------------------------------------------------------------
__global__ __launch_bounds__(256) void mla_attn_kernel(
    const bf16* __restrict__ Qh,   // [NH, T, 576]
    const bf16* __restrict__ Kg,   // [T, 576]
    const bf16* __restrict__ Kt,   // [512, T]  (V^T)
    bf16* __restrict__ Y)          // [T, 8192]
{
    const float scale = 0.044194173824159216f;  // 1/sqrt(512)
    const int h    = blockIdx.y;
    const int q0   = blockIdx.x * 32;
    const int tid  = threadIdx.x;
    const int wave = tid >> 5;
    const int lane = tid & 31;
    const int hf   = lane >> 4;
    const int l16  = lane & 15;
    const int smt  = wave >> 2;   // 0..1 : S row tile
    const int snt  = wave & 3;    // 0..3 : S col tile

    __shared__ __align__(16) bf16  Ks[2][64 * QKHS];   // 2 x 72KB TDM tiles
    __shared__ __align__(16) float Ss[32 * 64];
    __shared__ __align__(16) bf16  Ps[32 * 64];
    __shared__ float row_m[32], row_l[32], row_a[32];

    if (tid < 32) { row_m[tid] = -1e30f; row_l[tid] = 0.0f; }

    v8f o[2][4];
#pragma unroll
    for (int i = 0; i < 2; ++i)
#pragma unroll
        for (int j = 0; j < 4; ++j) o[i][j] = v8f_zero();

    const bf16* Qbase = Qh + ((size_t)h * TT + q0) * QKHS;
    const int nkt = (q0 + 32 + 63) >> 6;   // causal key-tiles (last k0 <= q0)

    if (wave == 0) tdm_load_k_tile(Kg, &Ks[0][0]);   // tile 0

    for (int kt = 0; kt < nkt; ++kt) {
        const int k0  = kt * 64;
        const int buf = kt & 1;

        if (wave == 0) {
            if (kt + 1 < nkt) {
                tdm_load_k_tile(Kg + (size_t)(k0 + 64) * QKHS, &Ks[buf ^ 1][0]);
                __builtin_amdgcn_s_wait_tensorcnt(1);  // current tile done
            } else {
                __builtin_amdgcn_s_wait_tensorcnt(0);
            }
        }
        if (kt + 1 < nkt)  // warm next V^T stripe into cache
            __builtin_prefetch(Kt + (size_t)(tid << 1) * TT + k0 + 64, 0, 1);
        __syncthreads();   // publish Ks[buf] to all waves

        // ---- S = Q * K^T, one 16x16 subtile per wave, K depth 576 ----
        v8f sacc = v8f_zero();
        for (int ks = 0; ks < 18; ++ks) {
            Frag aq, bk;
            const bf16* ap = Qbase + (size_t)(smt * 16 + l16) * QKHS + ks * 32 + hf * 8;
            aq.u[0] = *(const u32x4*)ap;
            aq.u[1] = *(const u32x4*)(ap + 16);
            const bf16* bp = &Ks[buf][(snt * 16 + l16) * QKHS + ks * 32 + hf * 8];
            bk.u[0] = *(const u32x4*)bp;
            bk.u[1] = *(const u32x4*)(bp + 16);
            sacc = __builtin_amdgcn_wmma_f32_16x16x32_bf16(
                false, aq.v, false, bk.v, (short)0, sacc, false, false);
        }
        {
            int col   = snt * 16 + l16;
            int rbase = smt * 16 + hf * 8;
#pragma unroll
            for (int r = 0; r < 8; ++r) Ss[(rbase + r) * 64 + col] = sacc[r];
        }
        __syncthreads();

        // ---- online softmax: one thread per query row ----
        if (tid < 32) {
            int r = tid;
            int jmax = (q0 + r) - k0 + 1;
            if (jmax > 64) jmax = 64;
            float m_old = row_m[r];
            float mx = m_old;
            for (int j = 0; j < jmax; ++j) {
                float v = Ss[r * 64 + j] * scale;
                mx = v > mx ? v : mx;
            }
            float a = __expf(m_old - mx);
            float lsum = row_l[r] * a;
            for (int j = 0; j < 64; ++j) {
                float p = (j < jmax) ? __expf(Ss[r * 64 + j] * scale - mx) : 0.0f;
                Ps[r * 64 + j] = (bf16)p;
                lsum += p;
            }
            row_m[r] = mx; row_l[r] = lsum; row_a[r] = a;
        }
        __syncthreads();

        // ---- rescale O, accumulate P @ V (wave owns cols [wave*64, +64)) ----
#pragma unroll
        for (int mt = 0; mt < 2; ++mt) {
            float ar[8];
#pragma unroll
            for (int r = 0; r < 8; ++r) ar[r] = row_a[mt * 16 + hf * 8 + r];
#pragma unroll
            for (int nt = 0; nt < 4; ++nt)
#pragma unroll
                for (int r = 0; r < 8; ++r) o[mt][nt][r] *= ar[r];
        }
#pragma unroll
        for (int kk = 0; kk < 2; ++kk) {
            Frag pf[2], vfg[4];
#pragma unroll
            for (int mt = 0; mt < 2; ++mt) {
                const bf16* pp = &Ps[(mt * 16 + l16) * 64 + kk * 32 + hf * 8];
                pf[mt].u[0] = *(const u32x4*)pp;
                pf[mt].u[1] = *(const u32x4*)(pp + 16);
            }
#pragma unroll
            for (int nt = 0; nt < 4; ++nt) {
                const bf16* vp = Kt + (size_t)(wave * 64 + nt * 16 + l16) * TT
                                    + k0 + kk * 32 + hf * 8;
                vfg[nt].u[0] = *(const u32x4*)vp;
                vfg[nt].u[1] = *(const u32x4*)(vp + 16);
            }
#pragma unroll
            for (int mt = 0; mt < 2; ++mt)
#pragma unroll
                for (int nt = 0; nt < 4; ++nt)
                    o[mt][nt] = __builtin_amdgcn_wmma_f32_16x16x32_bf16(
                        false, pf[mt].v, false, vfg[nt].v, (short)0, o[mt][nt], false, false);
        }
    }

    // ---- epilogue: divide by l, write bf16 y[t, h*512 + col] ----
#pragma unroll
    for (int mt = 0; mt < 2; ++mt) {
        float linv[8];
#pragma unroll
        for (int r = 0; r < 8; ++r) linv[r] = 1.0f / row_l[mt * 16 + hf * 8 + r];
#pragma unroll
        for (int nt = 0; nt < 4; ++nt) {
            int col = h * KVLOW + wave * 64 + nt * 16 + l16;
#pragma unroll
            for (int r = 0; r < 8; ++r) {
                int row = q0 + mt * 16 + hf * 8 + r;
                Y[(size_t)row * YDIM + col] = (bf16)(o[mt][nt][r] * linv[r]);
            }
        }
    }
}

// ---------------------------------------------------------------------------
// Host launch
// ---------------------------------------------------------------------------
extern "C" void kernel_launch(void* const* d_in, const int* in_sizes, int n_in,
                              void* d_out, int out_size, void* d_ws, size_t ws_size,
                              hipStream_t stream) {
    (void)in_sizes; (void)n_in; (void)out_size; (void)ws_size;
    const float* x     = (const float*)d_in[0];
    const float* cosT  = (const float*)d_in[1];
    const float* sinT  = (const float*)d_in[2];
    const float* Wqkv  = (const float*)d_in[3];
    const float* Wqdec = (const float*)d_in[4];
    const float* Wout  = (const float*)d_in[5];
    float* out = (float*)d_out;
    char*  ws  = (char*)d_ws;

    // Transient region (75.5 MB): [x_bf | lat_f] early, q_f later (disjoint lifetimes)
    bf16*  x_bf  = (bf16*)(ws + 0);
    float* lat_f = (float*)(ws + 8388608);
    float* q_f   = (float*)(ws + 0);
    size_t p = 75497472;
    bf16* wqkvT  = (bf16*)(ws + p); p += (size_t)DEMB * NQKV * 2;       // [1600][2048]
    bf16* wqdecT = (bf16*)(ws + p); p += (size_t)QLOW * QDIM * 2;       // [9216][1024]
    bf16* woutT  = (bf16*)(ws + p); p += (size_t)YDIM * DEMB * 2;       // [2048][8192]
    bf16* k_bf   = (bf16*)(ws + p); p += (size_t)TT * QKHS * 2;         // [2048][576]
    bf16* kt_bf  = (bf16*)(ws + p); p += (size_t)KVLOW * TT * 2;        // [512][2048]
    bf16* cq_bf  = (bf16*)(ws + p); p += (size_t)TT * QLOW * 2;
    bf16* qh_bf  = (bf16*)(ws + p); p += (size_t)NHEAD * TT * QKHS * 2;
    bf16* y_bf   = (bf16*)(ws + p); p += (size_t)TT * YDIM * 2;

    // 1) convert inputs to bf16 (weights transposed for TN GEMM)
    cvt_bf16_kernel<<<2048, 256, 0, stream>>>(x, x_bf, TT * DEMB);
    cvt_bf16_t_kernel<<<dim3(NQKV / 32, DEMB / 32), 256, 0, stream>>>(Wqkv, wqkvT, DEMB, NQKV);
    cvt_bf16_t_kernel<<<dim3(QDIM / 32, QLOW / 32), 256, 0, stream>>>(Wqdec, wqdecT, QLOW, QDIM);
    cvt_bf16_t_kernel<<<dim3(DEMB / 32, YDIM / 32), 256, 0, stream>>>(Wout, woutT, YDIM, DEMB);

    // 2) latents = x @ W_qkv
    gemm_bf16_tn_kernel<<<dim3((NQKV + BN - 1) / BN, TT / BM), 256, 0, stream>>>(
        x_bf, wqkvT, lat_f, TT, NQKV, DEMB);

    // 3) split + RoPE(k_r); then V^T = K[:, :512]^T
    split_rope_kernel<<<TT, 256, 0, stream>>>(lat_f, cosT, sinT, k_bf, cq_bf);
    transpose_bf16_kernel<<<dim3(KVLOW / 32, TT / 32), 256, 0, stream>>>(k_bf, kt_bf, TT, QKHS);

    // 4) q = c_q @ W_qdec
    gemm_bf16_tn_kernel<<<dim3(QDIM / BN, TT / BM), 256, 0, stream>>>(
        cq_bf, wqdecT, q_f, TT, QDIM, QLOW);

    // 5) RoPE(q_r), repack head-major bf16
    q_rope_kernel<<<TT, 256, 0, stream>>>(q_f, cosT, sinT, qh_bf);

    // 6) flash causal attention -> y bf16
    mla_attn_kernel<<<dim3(TT / 32, NHEAD), 256, 0, stream>>>(qh_bf, k_bf, kt_bf, y_bf);

    // 7) out = y @ W_out
    gemm_bf16_tn_kernel<<<dim3(DEMB / BN, TT / BM), 256, 0, stream>>>(
        y_bf, woutT, out, TT, DEMB, YDIM);
}
